// MabGmlpBlock_27058293965218
// MI455X (gfx1250) — matrix operand order repaired and checked by reading, not compiled, not run on hardware
//
#include <hip/hip_runtime.h>
#include <math.h>

typedef __attribute__((ext_vector_type(16))) __bf16 v16bf;
typedef __attribute__((ext_vector_type(8)))  __bf16 v8bf;
typedef __attribute__((ext_vector_type(8)))  float  v8f;

#define BB 32
#define NN 196
#define CC 256
#define EC 1536
#define HH 768

__device__ __forceinline__ float gelu_exact(float x) {
    return 0.5f * x * (1.0f + erff(x * 0.7071067811865475f));
}

// Build a v16bf A-fragment from two contiguous 8-element bf16 runs.
__device__ __forceinline__ v16bf load_a16(const __bf16* p0, const __bf16* p1) {
    v8bf x0 = *(const v8bf*)p0;
    v8bf x1 = *(const v8bf*)p1;
    v16bf r;
#pragma unroll
    for (int i = 0; i < 8; ++i) { r[i] = x0[i]; r[8 + i] = x1[i]; }
    return r;
}

// Async-copy one 32-row x 16-col fp32 weight tile (2 KB) into this wave's LDS
// buffer: 4 x global_load_async_to_lds_b128, each 32 lanes x 16B = 8 rows.
// Tracked by ASYNCcnt.
__device__ __forceinline__ void issue_tile_async(const float* Wn, int kb, int colBase,
                                                 int lane, unsigned ldsByte, int ldn) {
    const int r0 = lane >> 2;            // 0..7 (row within 8-row slab)
    const int cf = (lane & 3) * 4;       // float offset within 16-col row
#pragma unroll
    for (int i = 0; i < 4; ++i) {
        const float* g = Wn + (size_t)(kb + i * 8 + r0) * ldn + colBase + cf;
        unsigned l = ldsByte + (unsigned)((i * 8 + r0) * 64 + (lane & 3) * 16);
        asm volatile("global_load_async_to_lds_b128 %0, %1, off"
                     :: "v"(l), "v"(g) : "memory");
    }
}

// Read a 32x16 B-fragment (col = lane&15, K = kb + 8*half + j (j<8), kb + 16 + 8*half + (j-8) (j>=8)) from the LDS
// staging buffer and convert fp32 -> bf16.
__device__ __forceinline__ v16bf lds_bfrag(const float* buf, int half, int m) {
    v16bf bm;
#pragma unroll
    for (int j = 0; j < 16; ++j) bm[j] = (__bf16)buf[(j < 8 ? half * 8 + j : 8 + half * 8 + j) * 16 + m];
    return bm;
}

// ---------------- Kernel 1: LN over C, fp32 -> bf16 -------------------------
__global__ __launch_bounds__(256) void k_ln1(const float* __restrict__ x,
                                             const float* __restrict__ g,
                                             const float* __restrict__ b,
                                             __bf16* __restrict__ xn) {
    const int row = blockIdx.x;            // b*N + n
    const int t = threadIdx.x;             // 0..255 == channel
    float val = x[(size_t)row * CC + t];
    float s = val, q = val * val;
#pragma unroll
    for (int o = 16; o > 0; o >>= 1) {
        s += __shfl_down(s, o, 32);
        q += __shfl_down(q, o, 32);
    }
    __shared__ float ss[8], qq[8], mv[2];
    const int wave = t >> 5, lane = t & 31;
    if (lane == 0) { ss[wave] = s; qq[wave] = q; }
    __syncthreads();
    if (t == 0) {
        float S = 0.f, Q = 0.f;
#pragma unroll
        for (int i = 0; i < 8; ++i) { S += ss[i]; Q += qq[i]; }
        float m = S * (1.0f / CC);
        float var = Q * (1.0f / CC) - m * m;
        mv[0] = m;
        mv[1] = rsqrtf(var + 1e-3f);
    }
    __syncthreads();
    float o = (val - mv[0]) * mv[1] * g[t] + b[t];
    xn[(size_t)row * CC + t] = (__bf16)o;
}

// ---------------- Kernel 2: GEMM1 [32x256]x[256x1536] + GELU, split u/v -----
// Weights streamed HBM -> LDS via async b128 copies, double buffered per wave.
__global__ __launch_bounds__(256) void k_gemm1(const __bf16* __restrict__ xn,
                                               const float* __restrict__ W1,
                                               const float* __restrict__ b1,
                                               float* __restrict__ uraw,
                                               float* __restrict__ vbuf) {
    __shared__ float tile[8 * 2 * 512];      // 8 waves x 2 bufs x 2KB
    const int n = blockIdx.x;
    const int wave = threadIdx.x >> 5;
    const int lane = threadIdx.x & 31;
    const int m = lane & 15, half = lane >> 4;
    const int colBase = blockIdx.y * 128 + wave * 16;
    const int col = colBase + m;             // e in [0,1536)

    const __bf16* A0 = xn + ((size_t)m * NN + n) * CC;        // batch rows 0..15
    const __bf16* A1 = xn + ((size_t)(m + 16) * NN + n) * CC; // batch rows 16..31
    const float* Wn = W1 + (size_t)n * CC * EC;
    float* wtile = tile + wave * 1024;
    const unsigned ldsByte = (unsigned)(uintptr_t)wtile;

    v8f acc0 = {}, acc1 = {};
    issue_tile_async(Wn, 0, colBase, lane, ldsByte, EC);
    for (int kb = 0, bs = 0; kb < CC; kb += 32, bs ^= 1) {
        if (kb + 32 < CC) {
            issue_tile_async(Wn, kb + 32, colBase, lane, ldsByte + (unsigned)((bs ^ 1) * 2048), EC);
            asm volatile("s_wait_asynccnt 0x4" ::: "memory");
        } else {
            asm volatile("s_wait_asynccnt 0x0" ::: "memory");
        }
        v16bf a0 = load_a16(A0 + kb + half * 8, A0 + kb + 16 + half * 8);
        v16bf a1 = load_a16(A1 + kb + half * 8, A1 + kb + 16 + half * 8);
        v16bf bm = lds_bfrag(wtile + bs * 512, half, m);
        acc0 = __builtin_amdgcn_wmma_f32_16x16x32_bf16(false, a0, false, bm,
                                                       (short)0, acc0, false, false);
        acc1 = __builtin_amdgcn_wmma_f32_16x16x32_bf16(false, a1, false, bm,
                                                       (short)0, acc1, false, false);
    }

    const float bias = b1[(size_t)n * EC + col];
    const bool isU = col < HH;
    const int ec = isU ? col : col - HH;
    float* dst = isU ? uraw : vbuf;
#pragma unroll
    for (int r = 0; r < 8; ++r) {
        const int b0 = r + 8 * half;
        dst[((size_t)b0 * NN + n) * HH + ec]        = gelu_exact(acc0[r] + bias);
        dst[((size_t)(b0 + 16) * NN + n) * HH + ec] = gelu_exact(acc1[r] + bias);
    }
}

// ---------------- Kernel 3: LN over H, write transposed [B,H,N] bf16 --------
__global__ __launch_bounds__(256) void k_ln2(const float* __restrict__ uraw,
                                             const float* __restrict__ g,
                                             const float* __restrict__ bta,
                                             __bf16* __restrict__ unt) {
    const int row = blockIdx.x;            // b*N + n
    const int b = row / NN, n = row % NN;
    const int t = threadIdx.x;
    const float* up = uraw + (size_t)row * HH;
    float v0 = up[t], v1 = up[t + 256], v2 = up[t + 512];
    float s = v0 + v1 + v2;
    float q = v0 * v0 + v1 * v1 + v2 * v2;
#pragma unroll
    for (int o = 16; o > 0; o >>= 1) {
        s += __shfl_down(s, o, 32);
        q += __shfl_down(q, o, 32);
    }
    __shared__ float ss[8], qq[8], mv[2];
    const int wave = t >> 5, lane = t & 31;
    if (lane == 0) { ss[wave] = s; qq[wave] = q; }
    __syncthreads();
    if (t == 0) {
        float S = 0.f, Q = 0.f;
#pragma unroll
        for (int i = 0; i < 8; ++i) { S += ss[i]; Q += qq[i]; }
        float m = S * (1.0f / HH);
        float var = Q * (1.0f / HH) - m * m;
        mv[0] = m;
        mv[1] = rsqrtf(var + 1e-3f);
    }
    __syncthreads();
    const float mean = mv[0], inv = mv[1];
    const size_t base = (size_t)b * HH * NN + n;
    unt[base + (size_t)t * NN]         = (__bf16)((v0 - mean) * inv * g[t] + bta[t]);
    unt[base + (size_t)(t + 256) * NN] = (__bf16)((v1 - mean) * inv * g[t + 256] + bta[t + 256]);
    unt[base + (size_t)(t + 512) * NN] = (__bf16)((v2 - mean) * inv * g[t + 512] + bta[t + 512]);
}

// ---------------- Kernel 4: GEMM2 per-channel [32x196]x[196x196], gate by v -
__global__ __launch_bounds__(256) void k_gemm2(const __bf16* __restrict__ unt,
                                               const float* __restrict__ W2,
                                               const float* __restrict__ b2,
                                               const float* __restrict__ vbuf,
                                               __bf16* __restrict__ z) {
    const int d = blockIdx.x;                 // channel 0..767
    const int wave = threadIdx.x >> 5;
    const int lane = threadIdx.x & 31;
    const int m = lane & 15, half = lane >> 4;
    const int ntile = blockIdx.y * 8 + wave;  // 0..15 (tiles >=13 produce zeros)
    const int col = ntile * 16 + m;           // output patch index
    const bool cok = col < NN;

    const __bf16* A0 = unt + ((size_t)m * HH + d) * NN;
    const __bf16* A1 = unt + ((size_t)(m + 16) * HH + d) * NN;
    const float* Wd = W2 + (size_t)d * NN * NN;

    v8f acc0 = {}, acc1 = {};
    for (int kb = 0; kb < NN; kb += 32) {     // 7 steps, last partial (guarded)
        v16bf a0, a1, bm;
#pragma unroll
        for (int j = 0; j < 8; ++j) {
            const int k0 = kb + half * 8 + j;
            const int k1 = kb + 16 + half * 8 + j;
            a0[j]     = (k0 < NN) ? A0[k0] : (__bf16)0.f;
            a0[8 + j] = (k1 < NN) ? A0[k1] : (__bf16)0.f;
            a1[j]     = (k0 < NN) ? A1[k0] : (__bf16)0.f;
            a1[8 + j] = (k1 < NN) ? A1[k1] : (__bf16)0.f;
        }
#pragma unroll
        for (int j = 0; j < 16; ++j) {
            const int k = kb + (j < 8 ? half * 8 + j : 8 + half * 8 + j);
            bm[j] = (k < NN && cok) ? (__bf16)Wd[(size_t)k * NN + col] : (__bf16)0.f;
        }
        acc0 = __builtin_amdgcn_wmma_f32_16x16x32_bf16(false, a0, false, bm,
                                                       (short)0, acc0, false, false);
        acc1 = __builtin_amdgcn_wmma_f32_16x16x32_bf16(false, a1, false, bm,
                                                       (short)0, acc1, false, false);
    }

    if (cok) {
        const float bias = b2[(size_t)d * NN + col];
#pragma unroll
        for (int r = 0; r < 8; ++r) {
            const int b0 = r + 8 * half;
            const size_t i0 = ((size_t)b0 * NN + col) * HH + d;
            const size_t i1 = ((size_t)(b0 + 16) * NN + col) * HH + d;
            z[i0] = (__bf16)((acc0[r] + bias) * vbuf[i0]);
            z[i1] = (__bf16)((acc1[r] + bias) * vbuf[i1]);
        }
    }
}

// ---------------- Kernel 5: GEMM3 [32x768]x[768x256] + bias + residual ------
// Same async-LDS weight streaming as GEMM1.
__global__ __launch_bounds__(256) void k_gemm3(const __bf16* __restrict__ z,
                                               const float* __restrict__ W3,
                                               const float* __restrict__ b3,
                                               const float* __restrict__ x,
                                               float* __restrict__ out) {
    __shared__ float tile[8 * 2 * 512];      // 8 waves x 2 bufs x 2KB
    const int n = blockIdx.x;
    const int wave = threadIdx.x >> 5;
    const int lane = threadIdx.x & 31;
    const int m = lane & 15, half = lane >> 4;
    const int colBase = blockIdx.y * 128 + wave * 16;
    const int col = colBase + m;             // c in [0,256)

    const __bf16* A0 = z + ((size_t)m * NN + n) * HH;
    const __bf16* A1 = z + ((size_t)(m + 16) * NN + n) * HH;
    const float* Wn = W3 + (size_t)n * HH * CC;
    float* wtile = tile + wave * 1024;
    const unsigned ldsByte = (unsigned)(uintptr_t)wtile;

    v8f acc0 = {}, acc1 = {};
    issue_tile_async(Wn, 0, colBase, lane, ldsByte, CC);
    for (int kb = 0, bs = 0; kb < HH; kb += 32, bs ^= 1) {   // 24 steps
        if (kb + 32 < HH) {
            issue_tile_async(Wn, kb + 32, colBase, lane, ldsByte + (unsigned)((bs ^ 1) * 2048), CC);
            asm volatile("s_wait_asynccnt 0x4" ::: "memory");
        } else {
            asm volatile("s_wait_asynccnt 0x0" ::: "memory");
        }
        v16bf a0 = load_a16(A0 + kb + half * 8, A0 + kb + 16 + half * 8);
        v16bf a1 = load_a16(A1 + kb + half * 8, A1 + kb + 16 + half * 8);
        v16bf bm = lds_bfrag(wtile + bs * 512, half, m);
        acc0 = __builtin_amdgcn_wmma_f32_16x16x32_bf16(false, a0, false, bm,
                                                       (short)0, acc0, false, false);
        acc1 = __builtin_amdgcn_wmma_f32_16x16x32_bf16(false, a1, false, bm,
                                                       (short)0, acc1, false, false);
    }

    const float bias = b3[(size_t)n * CC + col];
#pragma unroll
    for (int r = 0; r < 8; ++r) {
        const int b0 = r + 8 * half;
        const size_t i0 = ((size_t)b0 * NN + n) * CC + col;
        const size_t i1 = ((size_t)(b0 + 16) * NN + n) * CC + col;
        out[i0] = acc0[r] + bias + x[i0];
        out[i1] = acc1[r] + bias + x[i1];
    }
}

extern "C" void kernel_launch(void* const* d_in, const int* in_sizes, int n_in,
                              void* d_out, int out_size, void* d_ws, size_t ws_size,
                              hipStream_t stream) {
    (void)in_sizes; (void)n_in; (void)out_size; (void)ws_size;
    const float* x     = (const float*)d_in[0];
    const float* ln1_g = (const float*)d_in[1];
    const float* ln1_b = (const float*)d_in[2];
    const float* W1    = (const float*)d_in[3];
    const float* b1    = (const float*)d_in[4];
    const float* ln2_g = (const float*)d_in[5];
    const float* ln2_b = (const float*)d_in[6];
    const float* W2    = (const float*)d_in[7];
    const float* b2    = (const float*)d_in[8];
    const float* W3    = (const float*)d_in[9];
    const float* b3    = (const float*)d_in[10];
    float* out = (float*)d_out;

    // Workspace layout (256B aligned sub-buffers, z aliases uraw after LN2):
    //   xn   : bf16 [32,196,256]  3,211,264 B
    //   uraw : f32  [32,196,768] 19,267,584 B   (later reused as z bf16)
    //   vbuf : f32  [32,196,768] 19,267,584 B
    //   unt  : bf16 [32,768,196]  9,633,792 B
    char* ws = (char*)d_ws;
    __bf16* xn  = (__bf16*)(ws);
    float* uraw = (float*)(ws + 3211264);
    float* vbuf = (float*)(ws + 3211264 + 19267584);
    __bf16* unt = (__bf16*)(ws + 3211264 + 2 * 19267584);
    __bf16* zb  = (__bf16*)(ws + 3211264);   // alias over uraw (uraw dead after k_ln2)

    k_ln1 <<<dim3(BB * NN), 256, 0, stream>>>(x, ln1_g, ln1_b, xn);
    k_gemm1<<<dim3(NN, EC / 128), 256, 0, stream>>>(xn, W1, b1, uraw, vbuf);
    k_ln2 <<<dim3(BB * NN), 256, 0, stream>>>(uraw, ln2_g, ln2_b, unt);
    k_gemm2<<<dim3(HH, 2), 256, 0, stream>>>(unt, W2, b2, vbuf, zb);
    k_gemm3<<<dim3(NN, 2), 256, 0, stream>>>(zb, W3, b3, x, out);
}
